// DecoderKIDProMP_17446157156920
// MI455X (gfx1250) — matrix operand order repaired and checked by reading, not compiled, not run on hardware
//
#include <hip/hip_runtime.h>

typedef _Float16 f16;
typedef __attribute__((ext_vector_type(16))) _Float16 v16h;
typedef __attribute__((ext_vector_type(8)))  float    v8f;

#define WAVES 8
#define TPB   (WAVES * 32)
#define HID   256
#define LAT   16
#define NJ    7
#define H1S   264       // padded activation row stride (halves) -> 528B, 16B aligned
#define GRID  512       // persistent blocks; each wave grid-strides over 16-row tiles

static __device__ __forceinline__ v8f wmma_f16(v16h a, v16h b, v8f c) {
    return __builtin_amdgcn_wmma_f32_16x16x32_f16(false, a, false, b, (short)0, c, false, false);
}

__global__ __launch_bounds__(TPB) void decoder_fk_kernel(
    const float* __restrict__ latent, const float* __restrict__ tim,
    const float* __restrict__ W1, const float* __restrict__ b1,
    const float* __restrict__ W2, const float* __restrict__ b2,
    const float* __restrict__ W3, const float* __restrict__ b3,
    float* __restrict__ out, int Bn)
{
    // Weights pre-swizzled into WMMA B-fragment order: frag[(nt*KT+kt)*32 + lane][16]
    __shared__ __align__(16) f16 W1f[16 * 1 * 32 * 16];   //  16 KB (K padded 17->32)
    __shared__ __align__(16) f16 W2f[16 * 8 * 32 * 16];   // 128 KB
    __shared__ __align__(16) f16 W3f[ 1 * 8 * 32 * 16];   //   8 KB (N padded 7->16)
    __shared__ __align__(16) f16 h1s[WAVES][16 * H1S];    //  66 KB
    __shared__ __align__(16) f16 h2s[WAVES][16 * H1S];    //  66 KB
    __shared__ float b1s[HID], b2s[HID], b3s[16];
    __shared__ float qs[WAVES][16 * 8];

    const int tid  = threadIdx.x;
    const int wave = tid >> 5;
    const int lane = tid & 31;
    const int nn   = lane & 15;   // B/C column role, A row role
    const int hi   = lane >> 4;

    // ---- Stage weights into LDS once per block (coalesced global reads, f32 -> f16) ----
    for (int g = tid; g < 32 * HID; g += TPB) {           // W1: [17->32][256]
        int kr = g >> 8, nc = g & 255;
        f16 v = (kr < LAT + 1) ? (f16)W1[kr * HID + nc] : (f16)0.f;
        int khi = (kr >> 3) & 1, ki = (kr & 7) | ((kr >> 1) & 8), nt = nc >> 4, nl = nc & 15;
        W1f[((nt) * 32 + khi * 16 + nl) * 16 + ki] = v;
    }
    for (int g = tid; g < HID * HID; g += TPB) {          // W2: [256][256]
        int kr = g >> 8, nc = g & 255;
        int kt = kr >> 5, khi = (kr >> 3) & 1, ki = (kr & 7) | ((kr >> 1) & 8);
        int nt = nc >> 4, nl = nc & 15;
        W2f[((nt * 8 + kt) * 32 + khi * 16 + nl) * 16 + ki] = (f16)W2[g];
    }
    for (int g = tid; g < HID * 16; g += TPB) {           // W3: [256][7->16]
        int kr = g >> 4, nc = g & 15;
        f16 v = (nc < NJ) ? (f16)W3[kr * NJ + nc] : (f16)0.f;
        int kt = kr >> 5, khi = (kr >> 3) & 1, ki = (kr & 7) | ((kr >> 1) & 8);
        W3f[(kt * 32 + khi * 16 + nc) * 16 + ki] = v;
    }
    for (int g = tid; g < HID; g += TPB) { b1s[g] = b1[g]; b2s[g] = b2[g]; }
    if (tid < 16) b3s[tid] = (tid < NJ) ? b3[tid] : 0.f;
    __syncthreads();

    f16* h1w = &h1s[wave][0];
    f16* h2w = &h2s[wave][0];
    const int nWaveTiles = (Bn + 15) >> 4;                // 16-row tiles
    const int tileStride = gridDim.x * WAVES;

    // Persistent grid-stride loop: weights stay resident in LDS across tiles.
    for (int wt = blockIdx.x * WAVES + wave; wt < nWaveTiles; wt += tileStride) {
        const long long row0 = (long long)wt * 16;
        long long rA = row0 + nn;                 // this lane's A-matrix row
        if (rA >= Bn) rA = Bn - 1;                // clamp (keeps EXEC all-ones for WMMA)

        // Prefetch next tile's latent rows into cache while this tile computes.
        {
            long long rN = (long long)(wt + tileStride) * 16 + nn;
            if (rN < Bn)
                __builtin_prefetch(latent + (size_t)rN * LAT + hi * 8, 0, 0);
        }

        // ---- Build x = concat(latent, time, pad) A-fragment (16x32 f16) ----
        v16h xa;
        {
            const float* lp = latent + (size_t)rA * LAT + hi * 8;
            #pragma unroll
            for (int j = 0; j < 8; ++j) xa[j] = (f16)lp[j];
            float t = tim[rA];
            xa[8] = (hi == 0) ? (f16)t : (f16)0.f;   // k==16 is the time column
            #pragma unroll
            for (int j = 9; j < 16; ++j) xa[j] = (f16)0.f;
        }

        // ---- Layer 1: h1 = relu(x @ W1 + b1) ----
        #pragma unroll
        for (int nt = 0; nt < 16; ++nt) {
            v16h bf = *(const v16h*)(W1f + (nt * 32 + lane) * 16);
            v8f c = {};
            c = wmma_f16(xa, bf, c);
            float bb = b1s[nt * 16 + nn];
            #pragma unroll
            for (int j = 0; j < 8; ++j) {
                float v = c[j] + bb;
                v = v > 0.f ? v : 0.f;
                h1w[(hi * 8 + j) * H1S + nt * 16 + nn] = (f16)v;
            }
        }
        asm volatile("s_wait_dscnt 0" ::: "memory");

        // ---- Layer 2: h2 = relu(h1 @ W2 + b2) ----
        {
            const f16* ar = h1w + nn * H1S;
            v16h afrag[8];
            #pragma unroll
            for (int kt = 0; kt < 8; ++kt) {
                #pragma unroll
                for (int j = 0; j < 8; ++j) {
                    afrag[kt][j]     = ar[kt * 32 + hi * 8 + j];
                    afrag[kt][j + 8] = ar[kt * 32 + 16 + hi * 8 + j];
                }
            }
            for (int nt = 0; nt < 16; ++nt) {
                v8f c = {};
                #pragma unroll
                for (int kt = 0; kt < 8; ++kt) {
                    v16h bf = *(const v16h*)(W2f + ((nt * 8 + kt) * 32 + lane) * 16);
                    c = wmma_f16(afrag[kt], bf, c);
                }
                float bb = b2s[nt * 16 + nn];
                #pragma unroll
                for (int j = 0; j < 8; ++j) {
                    float v = c[j] + bb;
                    v = v > 0.f ? v : 0.f;
                    h2w[(hi * 8 + j) * H1S + nt * 16 + nn] = (f16)v;
                }
            }
        }
        asm volatile("s_wait_dscnt 0" ::: "memory");

        // ---- Layer 3: q = h2 @ W3 + b3 (N padded to 16) ----
        {
            const f16* ar = h2w + nn * H1S;
            v8f c = {};
            #pragma unroll
            for (int kt = 0; kt < 8; ++kt) {
                v16h a;
                #pragma unroll
                for (int j = 0; j < 8; ++j) {
                    a[j]     = ar[kt * 32 + hi * 8 + j];
                    a[j + 8] = ar[kt * 32 + 16 + hi * 8 + j];
                }
                v16h bf = *(const v16h*)(W3f + (kt * 32 + lane) * 16);
                c = wmma_f16(a, bf, c);
            }
            if (nn < 8) {
                float bb = b3s[nn];
                #pragma unroll
                for (int j = 0; j < 8; ++j)
                    qs[wave][(hi * 8 + j) * 8 + nn] = c[j] + bb;
            }
        }
        asm volatile("s_wait_dscnt 0" ::: "memory");

        // ---- Forward kinematics: one batch row per lane (lanes 0..15) ----
        if (lane < 16) {
            long long rr = row0 + lane;
            if (rr < Bn) {
                const float* q = &qs[wave][lane * 8];
                const float DH_A[NJ]  = {0.f, 0.f, 0.f, 0.0825f, -0.0825f, 0.f, 0.088f};
                const float DH_D[NJ]  = {0.333f, 0.f, 0.316f, 0.f, 0.384f, 0.f, 0.f};
                const float DH_CA[NJ] = {1.f, 0.f, 0.f, 0.f, 0.f, 0.f, 0.f};
                const float DH_SA[NJ] = {0.f, -1.f, 1.f, 1.f, -1.f, 1.f, 1.f};
                // cum = 3x4 top of the 4x4 chain (implicit bottom row [0,0,0,1])
                float c00 = 1.f, c01 = 0.f, c02 = 0.f, c03 = 0.f;
                float c10 = 0.f, c11 = 1.f, c12 = 0.f, c13 = 0.f;
                float c20 = 0.f, c21 = 0.f, c22 = 1.f, c23 = 0.f;
                #pragma unroll
                for (int i = 0; i < NJ; ++i) {
                    float st, ct;
                    __sincosf(q[i], &st, &ct);
                    float ca = DH_CA[i], sa = DH_SA[i], aa = DH_A[i], dd = DH_D[i];
                    float t00 = ct,  t01 = -st * ca, t02 =  st * sa, t03 = aa * ct;
                    float t10 = st,  t11 =  ct * ca, t12 = -ct * sa, t13 = aa * st;
                    float            t21 =  sa,      t22 =  ca,      t23 = dd;
                    float n00 = c00 * t00 + c01 * t10;
                    float n01 = c00 * t01 + c01 * t11 + c02 * t21;
                    float n02 = c00 * t02 + c01 * t12 + c02 * t22;
                    float n03 = c00 * t03 + c01 * t13 + c02 * t23 + c03;
                    float n10 = c10 * t00 + c11 * t10;
                    float n11 = c10 * t01 + c11 * t11 + c12 * t21;
                    float n12 = c10 * t02 + c11 * t12 + c12 * t22;
                    float n13 = c10 * t03 + c11 * t13 + c12 * t23 + c13;
                    float n20 = c20 * t00 + c21 * t10;
                    float n21 = c20 * t01 + c21 * t11 + c22 * t21;
                    float n22 = c20 * t02 + c21 * t12 + c22 * t22;
                    float n23 = c20 * t03 + c21 * t13 + c22 * t23 + c23;
                    c00 = n00; c01 = n01; c02 = n02; c03 = n03;
                    c10 = n10; c11 = n11; c12 = n12; c13 = n13;
                    c20 = n20; c21 = n21; c22 = n22; c23 = n23;
                }
                float tr  = c00 + c11 + c22;
                float eta = 0.5f * sqrtf(1.f + tr);
                float d0 = c21 - c12, d1 = c02 - c20, d2 = c10 - c01;
                float s0 = (d0 > 0.f) ? 1.f : ((d0 < 0.f) ? -1.f : 0.f);
                float s1 = (d1 > 0.f) ? 1.f : ((d1 < 0.f) ? -1.f : 0.f);
                float s2 = (d2 > 0.f) ? 1.f : ((d2 < 0.f) ? -1.f : 0.f);
                float e0 = 0.5f * s0 * sqrtf(c00 - c11 - c22 + 1.f);
                float e1 = 0.5f * s1 * sqrtf(c11 - c22 - c00 + 1.f);
                float e2 = 0.5f * s2 * sqrtf(c22 - c00 - c11 + 1.f);
                float* o = out + (size_t)rr * NJ;
                o[0] = c03; o[1] = c13; o[2] = c23;
                o[3] = e0;  o[4] = e1;  o[5] = e2;  o[6] = eta;
            }
        }
    }
}

extern "C" void kernel_launch(void* const* d_in, const int* in_sizes, int n_in,
                              void* d_out, int out_size, void* d_ws, size_t ws_size,
                              hipStream_t stream) {
    const float* latent = (const float*)d_in[0];
    const float* tim    = (const float*)d_in[1];
    const float* W1     = (const float*)d_in[2];
    const float* b1     = (const float*)d_in[3];
    const float* W2     = (const float*)d_in[4];
    const float* b2     = (const float*)d_in[5];
    const float* W3     = (const float*)d_in[6];
    const float* b3     = (const float*)d_in[7];
    float* out = (float*)d_out;

    int Bn = in_sizes[0] / LAT;               // 131072
    int nWaveTiles = (Bn + 15) / 16;          // 8192
    int maxBlocks = (nWaveTiles + WAVES - 1) / WAVES;
    int blocks = maxBlocks < GRID ? maxBlocks : GRID;
    decoder_fk_kernel<<<blocks, TPB, 0, stream>>>(latent, tim, W1, b1, W2, b2,
                                                  W3, b3, out, Bn);
}
